// Node_cls_Module_75788992905469
// MI455X (gfx1250) — compile-verified, matrix operand order checked by reading, not run on hardware
//
#include <hip/hip_runtime.h>
#include <cstdint>

// ---------------------------------------------------------------------------
// GAT (2 layers, heads=2, concat=False) + linear node classifier for gfx1250.
// Dense GEMMs run on the WMMA pipe (v_wmma_f32_16x16x32_f16, f32 accumulate)
// with operands pre-packed into the CDNA5 wave32 fragment layout.
// GEMM v2: one wave per 16-row strip; A fragments (all K) are held in
// registers and reused across all 16 column tiles -> 16x less A traffic.
// ---------------------------------------------------------------------------

typedef __attribute__((ext_vector_type(16))) _Float16 v16h;
typedef __attribute__((ext_vector_type(8)))  float    v8f;

#define IN_CH   128
#define HID     128
#define HEADS   2
#define NCLS    7
#define KDIM    128            // K of both GAT GEMMs
#define NCOL    (HEADS * HID)  // 256 output columns of both GAT GEMMs
#define KSTEPS  (KDIM / 32)    // 4 WMMA K-slabs
#define NTILES  (NCOL / 16)    // 16 column tiles per row strip
#define NEG_SLOPE 0.2f

// K swizzle inside a 32-wide slab for a 16-bit A/B fragment element
// (CDNA5 ISA "16-bit A-Matrix 16x32": VGPR j holds K pairs; lanes 16-31 get +8).
__device__ __forceinline__ int kswz(int j, int h, int hi) {
  return (j < 4 ? 2 * j : 16 + 2 * (j - 4)) + h + (hi ? 8 : 0);
}

// ---------------- fragment packing (f32 -> f16, swizzled) -------------------

// A: X[M x 128] row-major f32 -> PA halves, layout ((mt*KSTEPS+ks)*32+lane)*16+e
__global__ void pack_a_wmma(const float* __restrict__ X, _Float16* __restrict__ PA, int M) {
  int idx = blockIdx.x * blockDim.x + threadIdx.x;
  if (idx >= M * KDIM) return;
  int e    = idx & 15;
  int lane = (idx >> 4) & 31;
  int t    = idx >> 9;                 // mt*KSTEPS + ks
  int mt = t / KSTEPS, ks = t % KSTEPS;
  int hi = lane >> 4, r = lane & 15;
  int k  = ks * 32 + kswz(e >> 1, e & 1, hi);
  PA[idx] = (_Float16)X[(size_t)(mt * 16 + r) * KDIM + k];
}

// B: W[128 x 256] row-major f32 -> PB halves, layout ((nt*KSTEPS+ks)*32+lane)*16+e
__global__ void pack_b_wmma(const float* __restrict__ W, _Float16* __restrict__ PB) {
  int idx = blockIdx.x * blockDim.x + threadIdx.x;
  if (idx >= KDIM * NCOL) return;
  int e    = idx & 15;
  int lane = (idx >> 4) & 31;
  int t    = idx >> 9;                 // nt*KSTEPS + ks
  int nt = t / KSTEPS, ks = t % KSTEPS;
  int hi = lane >> 4, r = lane & 15;
  int k  = ks * 32 + kswz(e >> 1, e & 1, hi);
  PB[idx] = (_Float16)W[(size_t)k * NCOL + nt * 16 + r];
}

// ------------------------- WMMA GEMM: Y = X @ W -----------------------------
// One wave per 16-row strip of Y (16 x 256). The strip's A fragments for all
// four K-slabs (4 x v16h = 32 VGPRs) are loaded once and reused across the 16
// column tiles; per tile: 8x global_load_b128 of B + 4x WMMA + 8 stores.
__global__ void gat_gemm_wmma(const _Float16* __restrict__ PA,
                              const _Float16* __restrict__ PB,
                              float* __restrict__ Y, int M) {
  const int lane = threadIdx.x & 31;
  const int mt   = (blockIdx.x * blockDim.x + threadIdx.x) >> 5;  // row strip
  if (mt * 16 >= M) return;            // wave-uniform; EXEC stays all-ones
  const int hi = lane >> 4, r = lane & 15;

  const v16h* pa = (const v16h*)PA + (size_t)mt * KSTEPS * 32 + lane;
  const v16h a0 = pa[0 * 32];
  const v16h a1 = pa[1 * 32];
  const v16h a2 = pa[2 * 32];
  const v16h a3 = pa[3 * 32];

  const v16h* pb  = (const v16h*)PB + lane;
  float* yrow = Y + (size_t)(mt * 16 + hi * 8) * NCOL + r;

#pragma unroll 4
  for (int nt = 0; nt < NTILES; ++nt) {
    const v16h* pbt = pb + (size_t)nt * KSTEPS * 32;
    v8f acc = {};
    acc = __builtin_amdgcn_wmma_f32_16x16x32_f16(false, a0, false, pbt[0 * 32],
                                                 (short)0, acc, false, false);
    acc = __builtin_amdgcn_wmma_f32_16x16x32_f16(false, a1, false, pbt[1 * 32],
                                                 (short)0, acc, false, false);
    acc = __builtin_amdgcn_wmma_f32_16x16x32_f16(false, a2, false, pbt[2 * 32],
                                                 (short)0, acc, false, false);
    acc = __builtin_amdgcn_wmma_f32_16x16x32_f16(false, a3, false, pbt[3 * 32],
                                                 (short)0, acc, false, false);
    // C/D layout: VGPR i -> M = i + 8*hi, N = lane&15
    float* yp = yrow + nt * 16;
#pragma unroll
    for (int i = 0; i < 8; ++i) yp[(size_t)i * NCOL] = acc[i];
  }
}

// ------------------- attention scalars: per-(node,head) dots ----------------
__global__ void alpha_kernel(const float* __restrict__ xh,
                             const float* __restrict__ aw_src,
                             const float* __restrict__ aw_dst,
                             float* __restrict__ asrc, float* __restrict__ adst, int N) {
  int idx = blockIdx.x * blockDim.x + threadIdx.x;   // n*HEADS + h
  if (idx >= N * HEADS) return;
  int hh = idx & (HEADS - 1), n = idx >> 1;
  const float4* v  = (const float4*)(xh + (size_t)n * NCOL + hh * HID);
  const float4* ss = (const float4*)(aw_src + hh * HID);
  const float4* dd = (const float4*)(aw_dst + hh * HID);
  float s = 0.f, d = 0.f;
#pragma unroll 8
  for (int c = 0; c < HID / 4; ++c) {
    float4 xv = v[c], av = ss[c], bv = dd[c];
    s += xv.x * av.x + xv.y * av.y + xv.z * av.z + xv.w * av.w;
    d += xv.x * bv.x + xv.y * bv.y + xv.z * bv.z + xv.w * bv.w;
  }
  asrc[idx] = s;
  adst[idx] = d;
}

// ---------------------------- edge-wise kernels -----------------------------
__device__ __forceinline__ void atomic_max_f32(float* addr, float val) {
  // float ordering via int (>=0) / uint (<0) punning; init value is -inf.
  if (val >= 0.0f) atomicMax((int*)addr, __float_as_int(val));
  else             atomicMin((unsigned int*)addr, __float_as_uint(val));
}

__device__ __forceinline__ float edge_logit(const float* asrc, const float* adst,
                                            int s, int d, int hh) {
  float v = asrc[s * HEADS + hh] + adst[d * HEADS + hh];
  return v > 0.0f ? v : NEG_SLOPE * v;               // leaky_relu(0.2)
}

__device__ __forceinline__ void edge_endpoints(const int* ei, int e, int E,
                                               int& s, int& d) {
  // e < E: real edge; e >= E: self loop on node (e - E)
  if (e < E) { s = ei[e]; d = ei[E + e]; }
  else       { s = e - E; d = e - E; }
}

__global__ void edge_max_kernel(const int* __restrict__ ei,
                                const float* __restrict__ asrc,
                                const float* __restrict__ adst,
                                float* __restrict__ m, int E, int N) {
  int idx = blockIdx.x * blockDim.x + threadIdx.x;   // (edge,head)
  if (idx >= (E + N) * HEADS) return;
  int hh = idx & (HEADS - 1), e = idx >> 1;
  int s, d; edge_endpoints(ei, e, E, s, d);
  atomic_max_f32(&m[d * HEADS + hh], edge_logit(asrc, adst, s, d, hh));
}

__global__ void edge_w_kernel(const int* __restrict__ ei,
                              const float* __restrict__ asrc,
                              const float* __restrict__ adst,
                              const float* __restrict__ m,
                              float* __restrict__ ewt, float* __restrict__ denom,
                              int E, int N) {
  int idx = blockIdx.x * blockDim.x + threadIdx.x;
  if (idx >= (E + N) * HEADS) return;
  int hh = idx & (HEADS - 1), e = idx >> 1;
  int s, d; edge_endpoints(ei, e, E, s, d);
  float mm = m[d * HEADS + hh];
  if (mm < -3.0e38f) mm = 0.0f;                      // where(isfinite(m), m, 0)
  float w = __expf(edge_logit(asrc, adst, s, d, hh) - mm);
  ewt[idx] = w;
  atomicAdd(&denom[d * HEADS + hh], w);
}

// One wave per (edge,head): lanes cover 128 channels as float4; L2-resident
// f32 atomic adds into the per-destination accumulator.
__global__ void aggregate_kernel(const int* __restrict__ ei,
                                 const float* __restrict__ xh,
                                 const float* __restrict__ ewt,
                                 const float* __restrict__ denom,
                                 float* __restrict__ acc, int E, int N) {
  int wave = (blockIdx.x * blockDim.x + threadIdx.x) >> 5;
  int lane = threadIdx.x & 31;
  if (wave >= (E + N) * HEADS) return;
  int hh = wave & (HEADS - 1), e = wave >> 1;
  int s, d; edge_endpoints(ei, e, E, s, d);
  float alpha = ewt[wave] / (denom[d * HEADS + hh] + 1e-16f);
  const float4* src = (const float4*)(xh + (size_t)s * NCOL + hh * HID);
  float*        dst = acc + (size_t)d * NCOL + hh * HID;
  float4 v = src[lane];
  atomicAdd(&dst[lane * 4 + 0], v.x * alpha);
  atomicAdd(&dst[lane * 4 + 1], v.y * alpha);
  atomicAdd(&dst[lane * 4 + 2], v.z * alpha);
  atomicAdd(&dst[lane * 4 + 3], v.w * alpha);
}

// head-mean + bias (+ optional relu)
__global__ void finalize_kernel(const float* __restrict__ acc,
                                const float* __restrict__ bias,
                                float* __restrict__ hout, int N, int do_relu) {
  int idx = blockIdx.x * blockDim.x + threadIdx.x;   // n*HID + c
  if (idx >= N * HID) return;
  int n = idx >> 7, c = idx & (HID - 1);
  float v = 0.5f * (acc[(size_t)n * NCOL + c] + acc[(size_t)n * NCOL + HID + c]) + bias[c];
  if (do_relu) v = fmaxf(v, 0.0f);
  hout[idx] = v;
}

// classifier: out[n,k] = h[n,:] @ Wc[:,k] + bc[k]   (7 cols -> plain VALU)
__global__ void classifier_kernel(const float* __restrict__ h,
                                  const float* __restrict__ Wc,
                                  const float* __restrict__ bc,
                                  float* __restrict__ out, int N) {
  int idx = blockIdx.x * blockDim.x + threadIdx.x;
  if (idx >= N * NCLS) return;
  int n = idx / NCLS, k = idx - n * NCLS;
  const float* hv = h + (size_t)n * HID;
  float s = bc[k];
#pragma unroll 8
  for (int c = 0; c < HID; ++c) s = fmaf(hv[c], Wc[c * NCLS + k], s);
  out[idx] = s;
}

__global__ void fill_kernel(float* __restrict__ p, float v, long long n) {
  long long i = blockIdx.x * (long long)blockDim.x + threadIdx.x;
  if (i < n) p[i] = v;
}

// ---------------------------------------------------------------------------
static inline unsigned cdiv(long long a, long long b) { return (unsigned)((a + b - 1) / b); }

static void run_gat_layer(const float* xin, const float* W, const float* aws,
                          const float* awd, const float* bias, const int* ei,
                          int N, int E, int do_relu,
                          float* xh, float* acc, float* hout,
                          float* asrc, float* adst, float* m, float* denom,
                          float* ewt, _Float16* PA, _Float16* PB,
                          hipStream_t stream) {
  const int T = (E + N) * HEADS;
  pack_a_wmma<<<cdiv((long long)N * KDIM, 256), 256, 0, stream>>>(xin, PA, N);
  pack_b_wmma<<<cdiv(KDIM * NCOL, 256), 256, 0, stream>>>(W, PB);
  {
    long long waves = (long long)(N + 15) / 16;      // one wave per row strip
    gat_gemm_wmma<<<cdiv(waves, 4), 128, 0, stream>>>(PA, PB, xh, N);
  }
  alpha_kernel<<<cdiv((long long)N * HEADS, 256), 256, 0, stream>>>(xh, aws, awd, asrc, adst, N);
  fill_kernel<<<cdiv((long long)N * HEADS, 256), 256, 0, stream>>>(m, -INFINITY, (long long)N * HEADS);
  fill_kernel<<<cdiv((long long)N * HEADS, 256), 256, 0, stream>>>(denom, 0.0f, (long long)N * HEADS);
  fill_kernel<<<cdiv((long long)N * NCOL, 256), 256, 0, stream>>>(acc, 0.0f, (long long)N * NCOL);
  edge_max_kernel<<<cdiv(T, 256), 256, 0, stream>>>(ei, asrc, adst, m, E, N);
  edge_w_kernel<<<cdiv(T, 256), 256, 0, stream>>>(ei, asrc, adst, m, ewt, denom, E, N);
  aggregate_kernel<<<cdiv((long long)T * 32, 256), 256, 0, stream>>>(ei, xh, ewt, denom, acc, E, N);
  finalize_kernel<<<cdiv((long long)N * HID, 256), 256, 0, stream>>>(acc, bias, hout, N, do_relu);
}

extern "C" void kernel_launch(void* const* d_in, const int* in_sizes, int n_in,
                              void* d_out, int out_size, void* d_ws, size_t ws_size,
                              hipStream_t stream) {
  (void)n_in; (void)out_size; (void)ws_size;
  const float* x   = (const float*)d_in[0];
  const int*   ei  = (const int*)d_in[1];     // [2,E]: row0=src, row1=dst
  // d_in[2] = batch: unused by the reference computation
  const float* W1  = (const float*)d_in[3];
  const float* a1s = (const float*)d_in[4];
  const float* a1d = (const float*)d_in[5];
  const float* b1  = (const float*)d_in[6];
  const float* W2  = (const float*)d_in[7];
  const float* a2s = (const float*)d_in[8];
  const float* a2d = (const float*)d_in[9];
  const float* b2  = (const float*)d_in[10];
  const float* Wc  = (const float*)d_in[11];
  const float* bc  = (const float*)d_in[12];
  float* out = (float*)d_out;

  const int N = in_sizes[0] / IN_CH;
  const int E = in_sizes[1] / 2;

  // workspace carve-up (float units, offsets kept 32B-aligned)
  float* ws = (float*)d_ws;
  size_t off = 0;
  float* xh    = ws + off; off += (size_t)N * NCOL;          // GEMM out [N,H,C]
  float* acc   = ws + off; off += (size_t)N * NCOL;          // segment-sum accum
  float* h     = ws + off; off += (size_t)N * HID;           // layer output
  float* asrc  = ws + off; off += ((size_t)N * HEADS + 15) & ~15ull;
  float* adst  = ws + off; off += ((size_t)N * HEADS + 15) & ~15ull;
  float* m     = ws + off; off += ((size_t)N * HEADS + 15) & ~15ull;
  float* denom = ws + off; off += ((size_t)N * HEADS + 15) & ~15ull;
  float* ewt   = ws + off; off += ((size_t)(E + N) * HEADS + 15) & ~15ull;
  _Float16* PA = (_Float16*)(ws + off); off += (size_t)N * KDIM / 2;     // packed A frags
  _Float16* PB = (_Float16*)(ws + off); off += (size_t)KDIM * NCOL / 2;  // packed B frags

  // layer 1: x -> h (relu)
  run_gat_layer(x, W1, a1s, a1d, b1, ei, N, E, /*relu=*/1,
                xh, acc, h, asrc, adst, m, denom, ewt, PA, PB, stream);
  // layer 2: h -> h (no relu)
  run_gat_layer(h, W2, a2s, a2d, b2, ei, N, E, /*relu=*/0,
                xh, acc, h, asrc, adst, m, denom, ewt, PA, PB, stream);
  // classifier
  classifier_kernel<<<cdiv((long long)N * NCLS, 256), 256, 0, stream>>>(h, Wc, bc, out, N);
}
